// TransformerMP_39977555591341
// MI455X (gfx1250) — matrix operand order repaired and checked by reading, not compiled, not run on hardware
//
#include <hip/hip_runtime.h>
#include <math.h>

// ---------------------------------------------------------------------------
// Model constants (match reference)
// ---------------------------------------------------------------------------
#define BB   2
#define SS   2048
#define DD   2048
#define NLAY 4
#define NH_  16
#define NKV_ 4
#define HD_  128
#define FH_  5504
#define VV   32000
#define NREP (NH_ / NKV_)
#define EPSf 1e-5f
#define THETAf 500000.0f
#define NTOK (BB * SS)          // 4096 rows

// ---------------------------------------------------------------------------
// WMMA vector types (CDNA5 / gfx1250, wave32)
// ---------------------------------------------------------------------------
typedef __attribute__((ext_vector_type(16))) __bf16 v16bf;
typedef __attribute__((ext_vector_type(8)))  __bf16 v8bf;
typedef __attribute__((ext_vector_type(8)))  float  v8f;
typedef int v4i_ __attribute__((vector_size(16)));   // b128 payload type

// ---------------------------------------------------------------------------
// gfx1250 async global->LDS path (ASYNCcnt), with sync fallback
// ---------------------------------------------------------------------------
#if __has_builtin(__builtin_amdgcn_global_load_async_to_lds_b128) && \
    __has_builtin(__builtin_amdgcn_s_wait_asynccnt)
#define HAVE_ASYNC_LDS 1
#else
#define HAVE_ASYNC_LDS 0
#endif

#define AS1 __attribute__((address_space(1)))
#define AS3 __attribute__((address_space(3)))

__device__ __forceinline__ void async_copy16(const void* g, void* l) {
#if HAVE_ASYNC_LDS
  void* gg = const_cast<void*>(g);
  __builtin_amdgcn_global_load_async_to_lds_b128((AS1 v4i_*)gg, (AS3 v4i_*)l,
                                                 /*offset*/0, /*cpol*/0);
#else
  (void)g; (void)l;
#endif
}

// ---------------------------------------------------------------------------
// Generic batched strided GEMM:  C[z] = A[z] (MxK, row-major, lda)
//                                     * B[z] (KxN, row-major ldb; or N-major
//                                             "transposed" storage if TRANSB)
// Per-z offsets: off = (z/zdiv)*outer + ((z%zdiv)/rep)*inner  (elements)
// Block tile 128(M) x 128(N), K-step 32. 256 threads = 8 waves (wave32).
// Wave tile 32x64 => 2x4 grid of v_wmma_f32_16x16x32_bf16 (8 WMMA/K-step).
// Non-TRANSB path: double-buffered GLOBAL_LOAD_ASYNC_TO_LDS_B128 pipeline.
// All problem dims here are multiples of 128/32 -> no edge guards.
// ---------------------------------------------------------------------------
template <bool TRANSB, bool OUT_BF16>
__global__ __launch_bounds__(256) void gemm_wmma_kernel(
    const __bf16* __restrict__ A, const __bf16* __restrict__ Bm,
    void* __restrict__ Cv,
    int M, int N, int K, int lda, int ldb, int ldc,
    long oA, long iA, long oB, long iB, long oC, long iC,
    int zdiv, int repB)
{
  const int z  = blockIdx.z;
  const int zb = z / zdiv;
  const int zh = z % zdiv;
  A  += (long)zb * oA + (long)zh * iA;
  Bm += (long)zb * oB + (long)(zh / repB) * iB;
  const long coff = (long)zb * oC + (long)zh * iC;

  __shared__ __bf16 As[2][128][40];   // 128x32 tile, +8 pad (16B-aligned rows)
  __shared__ __bf16 Bs[2][32][136];   // 32x128 tile, +8 pad

  const int tid   = threadIdx.x;
  const int lane  = tid & 31;
  const int wave  = tid >> 5;
  const int waveM = wave >> 1;     // 0..3
  const int waveN = wave & 1;      // 0..1
  const int half  = lane >> 4;     // 0/1
  const int l16   = lane & 15;

  const int m0 = blockIdx.y * 128;
  const int n0 = blockIdx.x * 128;

  v8f acc[2][4] = {};

  // A-tile map: 512 16B-chunks; thread handles idx = tid, tid+256
  const int ar0 = tid >> 2,          ac0 = (tid & 3) * 8;
  const int ar1 = (tid + 256) >> 2,  ac1 = ((tid + 256) & 3) * 8;
  // B-tile map (no-T): idx -> K row = idx>>4, N chunk = (idx&15)*8
  const int br0 = tid >> 4,          bc0 = (tid & 15) * 8;
  const int br1 = (tid + 256) >> 4,  bc1 = ((tid + 256) & 15) * 8;

  // ---- per-K-step compute from LDS buffer `cur` ----
  auto compute = [&](int cur) {
    // A 16x32 bf16 frag: lane = row M=l16; elems 0..7 = K(half*8..+7),
    //                    elems 8..15 = K(16+half*8..+7)
    v16bf afrag[2];
    for (int mi = 0; mi < 2; ++mi) {
      const int row = waveM * 32 + mi * 16 + l16;
      const v8bf lo = *reinterpret_cast<const v8bf*>(&As[cur][row][half * 8]);
      const v8bf hi = *reinterpret_cast<const v8bf*>(&As[cur][row][16 + half * 8]);
      for (int e = 0; e < 8; ++e) { afrag[mi][e] = lo[e]; afrag[mi][8 + e] = hi[e]; }
    }
    // B 32x16 bf16 frag: lane = col N=l16; elems 0..7 = K(half*8..+7),
    //                    elems 8..15 = K(16+half*8..+7)
    v16bf bfrag[4];
    for (int ni = 0; ni < 4; ++ni) {
      const int nb = waveN * 64 + ni * 16;
      for (int e = 0; e < 8; ++e) { bfrag[ni][e] = Bs[cur][half * 8 + e][nb + l16]; bfrag[ni][8 + e] = Bs[cur][16 + half * 8 + e][nb + l16]; }
    }
    for (int mi = 0; mi < 2; ++mi)
      for (int ni = 0; ni < 4; ++ni)
        acc[mi][ni] = __builtin_amdgcn_wmma_f32_16x16x32_bf16(
            false, afrag[mi], false, bfrag[ni],
            (short)0, acc[mi][ni], false, false);
  };

#if HAVE_ASYNC_LDS
  if constexpr (!TRANSB) {
    // ---- async double-buffered pipeline (4 async b128 per thread / tile) ----
    auto issue = [&](int k0, int buf) {
      async_copy16(&A[(long)(m0 + ar0) * lda + k0 + ac0], &As[buf][ar0][ac0]);
      async_copy16(&A[(long)(m0 + ar1) * lda + k0 + ac1], &As[buf][ar1][ac1]);
      async_copy16(&Bm[(long)(k0 + br0) * ldb + n0 + bc0], &Bs[buf][br0][bc0]);
      async_copy16(&Bm[(long)(k0 + br1) * ldb + n0 + bc1], &Bs[buf][br1][bc1]);
    };
    issue(0, 0);
    int it = 0;
    for (int k0 = 0; k0 < K; k0 += 32, ++it) {
      const int cur = it & 1;
      if (k0 + 32 < K) {
        issue(k0 + 32, cur ^ 1);
        __builtin_amdgcn_s_wait_asynccnt(4);  // in-order: tile `cur` landed
      } else {
        __builtin_amdgcn_s_wait_asynccnt(0);
      }
      __syncthreads();            // all threads' chunks of tile `cur` visible
      compute(cur);
      __syncthreads();            // done reading before next overwrite
    }
  } else
#endif
  {
    // ---- sync fallback: regs -> barrier -> LDS -> barrier -> compute ----
    for (int k0 = 0; k0 < K; k0 += 32) {
      v8bf aR0 = *reinterpret_cast<const v8bf*>(&A[(long)(m0 + ar0) * lda + k0 + ac0]);
      v8bf aR1 = *reinterpret_cast<const v8bf*>(&A[(long)(m0 + ar1) * lda + k0 + ac1]);
      v8bf bR[2];
      int  tn[2], tk[2];
      if (!TRANSB) {
        bR[0] = *reinterpret_cast<const v8bf*>(&Bm[(long)(k0 + br0) * ldb + n0 + bc0]);
        bR[1] = *reinterpret_cast<const v8bf*>(&Bm[(long)(k0 + br1) * ldb + n0 + bc1]);
      } else {
        // Bt stored [N][K]: 512 chunks, n = idx>>2, k chunk = (idx&3)*8
        for (int i = 0; i < 2; ++i) {
          int idx = tid + i * 256;
          tn[i] = idx >> 2;
          tk[i] = (idx & 3) * 8;
          bR[i] = *reinterpret_cast<const v8bf*>(
              &Bm[(long)(n0 + tn[i]) * ldb + k0 + tk[i]]);
        }
      }
      if (k0 + 32 < K)
        __builtin_prefetch(&A[(long)(m0 + ar0) * lda + k0 + 32 + ac0], 0, 1);

      __syncthreads();
      *reinterpret_cast<v8bf*>(&As[0][ar0][ac0]) = aR0;
      *reinterpret_cast<v8bf*>(&As[0][ar1][ac1]) = aR1;
      if (!TRANSB) {
        *reinterpret_cast<v8bf*>(&Bs[0][br0][bc0]) = bR[0];
        *reinterpret_cast<v8bf*>(&Bs[0][br1][bc1]) = bR[1];
      } else {
        for (int i = 0; i < 2; ++i)
          for (int j = 0; j < 8; ++j)
            Bs[0][tk[i] + j][tn[i]] = bR[i][j];   // transpose on LDS write
      }
      __syncthreads();
      compute(0);
    }
  }

  // ---- store C (16x16 f32 layout: N = l16, M = half*8 + r) ----
  for (int mi = 0; mi < 2; ++mi)
    for (int ni = 0; ni < 4; ++ni) {
      const int n  = n0 + waveN * 64 + ni * 16 + l16;
      const int mb = m0 + waveM * 32 + mi * 16 + half * 8;
      for (int r = 0; r < 8; ++r) {
        const long off = coff + (long)(mb + r) * ldc + n;
        if (OUT_BF16) ((__bf16*)Cv)[off] = (__bf16)acc[mi][ni][r];
        else          ((float*)Cv)[off]  = acc[mi][ni][r];
      }
    }
}

// ---------------------------------------------------------------------------
// Support kernels
// ---------------------------------------------------------------------------
__global__ void cast_f32_bf16_kernel(const float* __restrict__ x,
                                     __bf16* __restrict__ y, long n) {
  long i = (long)blockIdx.x * blockDim.x + threadIdx.x;
  long st = (long)gridDim.x * blockDim.x;
  for (; i < n; i += st) y[i] = (__bf16)x[i];
}

__global__ void embed_kernel(const int* __restrict__ tokens,
                             const float* __restrict__ emb,
                             float* __restrict__ h, long n) {
  long i = (long)blockIdx.x * blockDim.x + threadIdx.x;
  long st = (long)gridDim.x * blockDim.x;
  for (; i < n; i += st) {
    long row = i / DD, col = i % DD;
    h[i] = emb[(long)tokens[row] * DD + col];
  }
}

__global__ __launch_bounds__(256) void rmsnorm_kernel(
    const float* __restrict__ x, const float* __restrict__ w,
    __bf16* __restrict__ y, int D) {
  const long row = blockIdx.x;
  const float* xr = x + row * (long)D;
  __shared__ float sred[256];
  float ss = 0.f;
  for (int j = threadIdx.x; j < D; j += 256) { float v = xr[j]; ss += v * v; }
  sred[threadIdx.x] = ss; __syncthreads();
  for (int s = 128; s > 0; s >>= 1) {
    if (threadIdx.x < (unsigned)s) sred[threadIdx.x] += sred[threadIdx.x + s];
    __syncthreads();
  }
  const float inv = rsqrtf(sred[0] / (float)D + EPSf);
  for (int j = threadIdx.x; j < D; j += 256)
    y[row * (long)D + j] = (__bf16)(xr[j] * inv * w[j]);
}

// in-place RoPE on bf16 [B, S, nh, HD]; idx enumerates (b,s,h,i) pairs
__global__ void rope_kernel(__bf16* __restrict__ x, int nh, long npairs) {
  long i = (long)blockIdx.x * blockDim.x + threadIdx.x;
  long st = (long)gridDim.x * blockDim.x;
  const int hd2 = HD_ / 2;
  for (; i < npairs; i += st) {
    const int  fi  = (int)(i % hd2);
    const long rw  = i / hd2;                 // (b*S + s)*nh + h
    const int  pos = (int)((rw / nh) % SS);
    const long base = rw * HD_ + 2 * fi;
    const float freq = __expf(-((2.f * fi) / (float)HD_) * __logf(THETAf));
    const float ang  = (float)pos * freq;
    float sn, cs;
    __sincosf(ang, &sn, &cs);
    const float xr = (float)x[base], xi = (float)x[base + 1];
    x[base]     = (__bf16)(xr * cs - xi * sn);
    x[base + 1] = (__bf16)(xr * sn + xi * cs);
  }
}

// causal softmax over fp32 scores row -> bf16 probs.  grid: (S, B*NH)
__global__ __launch_bounds__(256) void softmax_kernel(
    const float* __restrict__ scores, __bf16* __restrict__ probs, float scale) {
  const long z = blockIdx.y;
  const int  q = blockIdx.x;
  const float* row = scores + ((long)z * SS + q) * (long)SS;
  __bf16*     prow = probs  + ((long)z * SS + q) * (long)SS;
  __shared__ float sred[256];
  const int tid = threadIdx.x;
  float vals[SS / 256];
  float vmax = -3.4e38f;
  for (int i = 0; i < SS / 256; ++i) {
    const int j = tid + i * 256;
    const float v = (j <= q) ? row[j] * scale : -3.4e38f;
    vals[i] = v; vmax = fmaxf(vmax, v);
  }
  sred[tid] = vmax; __syncthreads();
  for (int s = 128; s > 0; s >>= 1) {
    if (tid < s) sred[tid] = fmaxf(sred[tid], sred[tid + s]);
    __syncthreads();
  }
  vmax = sred[0]; __syncthreads();
  float sum = 0.f;
  for (int i = 0; i < SS / 256; ++i) {
    const int j = tid + i * 256;
    const float e = (j <= q) ? __expf(vals[i] - vmax) : 0.f;
    vals[i] = e; sum += e;
  }
  sred[tid] = sum; __syncthreads();
  for (int s = 128; s > 0; s >>= 1) {
    if (tid < s) sred[tid] += sred[tid + s];
    __syncthreads();
  }
  const float r = 1.f / sred[0];
  for (int i = 0; i < SS / 256; ++i)
    prow[tid + i * 256] = (__bf16)(vals[i] * r);
}

__global__ void silu_mul_kernel(const float* __restrict__ g,
                                const float* __restrict__ u,
                                __bf16* __restrict__ y, long n) {
  long i = (long)blockIdx.x * blockDim.x + threadIdx.x;
  long st = (long)gridDim.x * blockDim.x;
  for (; i < n; i += st) {
    const float gv = g[i];
    y[i] = (__bf16)((gv / (1.f + __expf(-gv))) * u[i]);
  }
}

__global__ void add_inplace_kernel(float* __restrict__ h,
                                   const float* __restrict__ t, long n) {
  long i = (long)blockIdx.x * blockDim.x + threadIdx.x;
  long st = (long)gridDim.x * blockDim.x;
  for (; i < n; i += st) h[i] += t[i];
}

// ---------------------------------------------------------------------------
// Host-side launch
// ---------------------------------------------------------------------------
static inline int ew_blocks(long n) {
  long b = (n + 255) / 256;
  return (int)(b > 8192 ? 8192 : b);
}

static void gemm_f32(const __bf16* A, const __bf16* B, float* C,
                     int M, int N, int K, hipStream_t s) {
  dim3 g(N / 128, M / 128, 1);
  gemm_wmma_kernel<false, false><<<g, 256, 0, s>>>(A, B, C, M, N, K, K, N, N,
                                                   0, 0, 0, 0, 0, 0, 1, 1);
}
static void gemm_bf16(const __bf16* A, const __bf16* B, __bf16* C,
                      int M, int N, int K, hipStream_t s) {
  dim3 g(N / 128, M / 128, 1);
  gemm_wmma_kernel<false, true><<<g, 256, 0, s>>>(A, B, C, M, N, K, K, N, N,
                                                  0, 0, 0, 0, 0, 0, 1, 1);
}

extern "C" void kernel_launch(void* const* d_in, const int* in_sizes, int n_in,
                              void* d_out, int out_size, void* d_ws, size_t ws_size,
                              hipStream_t stream) {
  (void)in_sizes; (void)n_in; (void)out_size; (void)ws_size;
  const int*   tokens  = (const int*)  d_in[0];
  /* start_pos (d_in[1]) == 0 in this workload */
  const float* tok_emb = (const float*)d_in[2];
  const float* wq      = (const float*)d_in[3];
  const float* wk      = (const float*)d_in[4];
  const float* wv      = (const float*)d_in[5];
  const float* wo      = (const float*)d_in[6];
  const float* w1      = (const float*)d_in[7];
  const float* w2      = (const float*)d_in[8];
  const float* w3      = (const float*)d_in[9];
  const float* anw     = (const float*)d_in[10];
  const float* fnw     = (const float*)d_in[11];
  const float* finw    = (const float*)d_in[12];
  const float* out_w   = (const float*)d_in[13];
  float* logits = (float*)d_out;

  // ---- workspace bump allocator ----
  char*  ws  = (char*)d_ws;
  size_t off = 0;
  auto alloc = [&](size_t bytes) -> void* {
    void* p = ws + off;
    off = (off + bytes + 255) & ~(size_t)255;
    return p;
  };
  const long nWq = (long)NLAY * DD * DD;
  const long nWk = (long)NLAY * DD * (NKV_ * HD_);
  const long nW1 = (long)NLAY * DD * FH_;
  const long nOw = (long)DD * VV;

  __bf16* wq_b = (__bf16*)alloc(nWq * 2);
  __bf16* wk_b = (__bf16*)alloc(nWk * 2);
  __bf16* wv_b = (__bf16*)alloc(nWk * 2);
  __bf16* wo_b = (__bf16*)alloc(nWq * 2);
  __bf16* w1_b = (__bf16*)alloc(nW1 * 2);
  __bf16* w2_b = (__bf16*)alloc(nW1 * 2);
  __bf16* w3_b = (__bf16*)alloc(nW1 * 2);
  __bf16* ow_b = (__bf16*)alloc(nOw * 2);

  float*  h    = (float*) alloc((long)NTOK * DD * 4);
  __bf16* xn   = (__bf16*)alloc((long)NTOK * DD * 2);
  __bf16* q    = (__bf16*)alloc((long)NTOK * NH_  * HD_ * 2);
  __bf16* k    = (__bf16*)alloc((long)NTOK * NKV_ * HD_ * 2);
  __bf16* v    = (__bf16*)alloc((long)NTOK * NKV_ * HD_ * 2);
  float*  scor = (float*) alloc((long)BB * NH_ * SS * SS * 4);
  __bf16* prob = (__bf16*)alloc((long)BB * NH_ * SS * SS * 2);
  __bf16* aout = (__bf16*)alloc((long)NTOK * DD * 2);
  float*  t0   = (float*) alloc((long)NTOK * FH_ * 4);
  float*  t1   = (float*) alloc((long)NTOK * FH_ * 4);
  __bf16* hb   = (__bf16*)alloc((long)NTOK * FH_ * 2);
  float*  tmp  = (float*) alloc((long)NTOK * DD * 4);

  // ---- cast weights fp32 -> bf16 (deterministic, each call) ----
  cast_f32_bf16_kernel<<<ew_blocks(nWq), 256, 0, stream>>>(wq, wq_b, nWq);
  cast_f32_bf16_kernel<<<ew_blocks(nWk), 256, 0, stream>>>(wk, wk_b, nWk);
  cast_f32_bf16_kernel<<<ew_blocks(nWk), 256, 0, stream>>>(wv, wv_b, nWk);
  cast_f32_bf16_kernel<<<ew_blocks(nWq), 256, 0, stream>>>(wo, wo_b, nWq);
  cast_f32_bf16_kernel<<<ew_blocks(nW1), 256, 0, stream>>>(w1, w1_b, nW1);
  cast_f32_bf16_kernel<<<ew_blocks(nW1), 256, 0, stream>>>(w2, w2_b, nW1);
  cast_f32_bf16_kernel<<<ew_blocks(nW1), 256, 0, stream>>>(w3, w3_b, nW1);
  cast_f32_bf16_kernel<<<ew_blocks(nOw), 256, 0, stream>>>(out_w, ow_b, nOw);

  // ---- embedding ----
  const long nH = (long)NTOK * DD;
  embed_kernel<<<ew_blocks(nH), 256, 0, stream>>>(tokens, tok_emb, h, nH);

  const int KV = NKV_ * HD_;   // 512
  const float scale = 1.0f / sqrtf((float)HD_);

  for (int l = 0; l < NLAY; ++l) {
    // ---- attention ----
    rmsnorm_kernel<<<NTOK, 256, 0, stream>>>(h, anw + (long)l * DD, xn, DD);
    gemm_bf16(xn, wq_b + (long)l * DD * DD, q, NTOK, DD, DD, stream);
    gemm_bf16(xn, wk_b + (long)l * DD * KV, k, NTOK, KV, DD, stream);
    gemm_bf16(xn, wv_b + (long)l * DD * KV, v, NTOK, KV, DD, stream);

    rope_kernel<<<ew_blocks((long)NTOK * NH_  * HD_ / 2), 256, 0, stream>>>(
        q, NH_, (long)NTOK * NH_ * HD_ / 2);
    rope_kernel<<<ew_blocks((long)NTOK * NKV_ * HD_ / 2), 256, 0, stream>>>(
        k, NKV_, (long)NTOK * NKV_ * HD_ / 2);

    // scores[z] = Q_z (SxHD) * K_z^T, z = b*NH + hq, GQA via repB
    {
      dim3 g(SS / 128, SS / 128, BB * NH_);
      gemm_wmma_kernel<true, false><<<g, 256, 0, stream>>>(
          q, k, scor, SS, SS, HD_,
          /*lda*/ NH_ * HD_, /*ldb*/ KV, /*ldc*/ SS,
          /*oA*/ (long)SS * NH_ * HD_, /*iA*/ HD_,
          /*oB*/ (long)SS * KV,        /*iB*/ HD_,
          /*oC*/ (long)NH_ * SS * SS,  /*iC*/ (long)SS * SS,
          NH_, NREP);
    }
    {
      dim3 g(SS, BB * NH_, 1);
      softmax_kernel<<<g, 256, 0, stream>>>(scor, prob, scale);
    }
    // attn_out[z] = probs_z (SxS) * V_z (SxHD)
    {
      dim3 g(HD_ / 128, SS / 128, BB * NH_);
      gemm_wmma_kernel<false, true><<<g, 256, 0, stream>>>(
          prob, v, aout, SS, HD_, SS,
          /*lda*/ SS, /*ldb*/ KV, /*ldc*/ NH_ * HD_,
          /*oA*/ (long)NH_ * SS * SS, /*iA*/ (long)SS * SS,
          /*oB*/ (long)SS * KV,       /*iB*/ HD_,
          /*oC*/ (long)SS * NH_ * HD_,/*iC*/ HD_,
          NH_, NREP);
    }
    gemm_f32(aout, wo_b + (long)l * DD * DD, tmp, NTOK, DD, DD, stream);
    add_inplace_kernel<<<ew_blocks(nH), 256, 0, stream>>>(h, tmp, nH);

    // ---- feed-forward ----
    rmsnorm_kernel<<<NTOK, 256, 0, stream>>>(h, fnw + (long)l * DD, xn, DD);
    gemm_f32(xn, w1_b + (long)l * DD * FH_, t0, NTOK, FH_, DD, stream);
    gemm_f32(xn, w3_b + (long)l * DD * FH_, t1, NTOK, FH_, DD, stream);
    silu_mul_kernel<<<ew_blocks((long)NTOK * FH_), 256, 0, stream>>>(
        t0, t1, hb, (long)NTOK * FH_);
    gemm_f32(hb, w2_b + (long)l * FH_ * DD, tmp, NTOK, DD, FH_, stream);
    add_inplace_kernel<<<ew_blocks(nH), 256, 0, stream>>>(h, tmp, nH);
  }

  // ---- final norm + output projection ----
  rmsnorm_kernel<<<NTOK, 256, 0, stream>>>(h, finw, xn, DD);
  gemm_f32(xn, ow_b, logits, NTOK, VV, DD, stream);
}
